// StandardAttention_87565793230943
// MI455X (gfx1250) — matrix operand order repaired and checked by reading, not compiled, not run on hardware
//
#include <hip/hip_runtime.h>

#define D_MODEL 1024
#define N_HEADS 16
#define HEAD_DIM 64
#define BATCH 2
#define SEQ 2048

typedef __bf16 bf16;
typedef __attribute__((ext_vector_type(16))) bf16  v16bf;
typedef __attribute__((ext_vector_type(8)))  bf16  v8bf;
typedef __attribute__((ext_vector_type(8)))  float v8f;

// ---------------------------------------------------------------- convert
__global__ void cvt_f32_bf16(const float* __restrict__ in,
                             bf16* __restrict__ out, int n) {
  int i = blockIdx.x * blockDim.x + threadIdx.x;
  int stride = gridDim.x * blockDim.x;
  for (; i < n; i += stride) out[i] = (bf16)in[i];
}

// ---------------------------------------------------------------- GEMM
// C[m,n] = sum_k A[m,k] * W[n,k]   (y = A @ W^T, torch Linear weights)
// One wave computes a 64x64 C macro-tile: 4x4 grid of 16x16 WMMA tiles.
// Per K=32 step: 4 A-frags + 4 B-frags (8KB) -> 16 WMMAs (32 FLOP/B).
// mode 0: bf16 out row-major [M,N]
// mode 1: bf16 out, V-transposed layout [B][H][HEAD_DIM][SEQ]
// mode 2: f32 out row-major [M,N]
__global__ __launch_bounds__(128)
void gemm_bf16_wmma(const bf16* __restrict__ A, const bf16* __restrict__ W,
                    void* __restrict__ out, int M, int N, int K, int mode) {
  const int lane = threadIdx.x & 31;
  const int warp = threadIdx.x >> 5;
  const int wid  = blockIdx.x * (blockDim.x >> 5) + warp;
  const int ntn  = N >> 6;                 // 64-wide N tiles
  const int tM   = wid / ntn;
  const int tN   = wid % ntn;
  if (tM >= (M >> 6)) return;

  const int row = lane & 15;
  const int loA = (lane & 16) ? 16 : 0;    // A-fragment K half
  const int loB = (lane & 16) ? 16 : 0;    // B-fragment K half
  const bf16* arow[4];
  const bf16* brow[4];
#pragma unroll
  for (int i = 0; i < 4; ++i) {
    arow[i] = A + (size_t)(tM * 64 + i * 16 + row) * K;
    brow[i] = W + (size_t)(tN * 64 + i * 16 + row) * K;
  }

  v8f acc[4][4];
#pragma unroll
  for (int i = 0; i < 4; ++i)
#pragma unroll
    for (int j = 0; j < 4; ++j) acc[i][j] = (v8f){};

  for (int k = 0; k < K; k += 32) {
    v16bf af[4], bf_[4];
#pragma unroll
    for (int i = 0; i < 4; ++i) {
      v8bf a0 = *(const v8bf*)(arow[i] + k + loA);
      v8bf a1 = *(const v8bf*)(arow[i] + k + loA + 8);
#pragma unroll
      for (int e = 0; e < 8; ++e) { af[i][e] = a0[e]; af[i][8 + e] = a1[e]; }
      bf_[i] = *(const v16bf*)(brow[i] + k + loB);
    }
#pragma unroll
    for (int i = 0; i < 4; ++i)
#pragma unroll
      for (int j = 0; j < 4; ++j)
        acc[i][j] = __builtin_amdgcn_wmma_f32_16x16x32_bf16(
            false, af[i], false, bf_[j], (short)0, acc[i][j], false, false);
  }

  const int hi8 = (lane & 16) ? 8 : 0;
#pragma unroll
  for (int i = 0; i < 4; ++i) {
    const int cm0 = tM * 64 + i * 16 + hi8;
#pragma unroll
    for (int j = 0; j < 4; ++j) {
      const int cn = tN * 64 + j * 16 + row;
      if (mode == 0) {
        bf16* o = (bf16*)out;
#pragma unroll
        for (int r = 0; r < 8; ++r)
          o[(size_t)(cm0 + r) * N + cn] = (bf16)acc[i][j][r];
      } else if (mode == 1) {
        bf16* o = (bf16*)out;
        const int h = cn >> 6, d = cn & 63;
#pragma unroll
        for (int r = 0; r < 8; ++r) {
          int m = cm0 + r;
          int b = m >> 11, s = m & (SEQ - 1);
          o[(((size_t)(b * N_HEADS + h) * HEAD_DIM + d) * SEQ) + s] =
              (bf16)acc[i][j][r];
        }
      } else {
        float* o = (float*)out;
#pragma unroll
        for (int r = 0; r < 8; ++r)
          o[(size_t)(cm0 + r) * N + cn] = acc[i][j][r];
      }
    }
  }
}

// ---------------------------------------------------------------- attention
// One wave per (b, h, 16-query tile). Flash-style streaming over 32-key blocks.
__global__ __launch_bounds__(32)
void attn_wmma(const bf16* __restrict__ Q, const bf16* __restrict__ Kc,
               const bf16* __restrict__ Vt, bf16* __restrict__ Ctx) {
  __shared__ bf16 pbuf[16 * 32];  // P tile bounce (C-layout -> A-layout)

  const int lane = threadIdx.x & 31;
  const int wid  = blockIdx.x;
  const int qtiles = SEQ / 16;                    // 128
  const int qt = wid % qtiles;
  const int h  = (wid / qtiles) % N_HEADS;
  const int b  = wid / (qtiles * N_HEADS);

  const int row = lane & 15;
  const int loA = (lane & 16) ? 16 : 0;
  const int loB = (lane & 16) ? 16 : 0;
  const int hi8 = (lane & 16) ? 8 : 0;

  // Q A-fragments: d = 0..31 (qa0), d = 32..63 (qa1)
  const bf16* qrow =
      Q + (size_t)(b * SEQ + qt * 16 + row) * D_MODEL + h * HEAD_DIM;
  v16bf qa0, qa1;
  {
    v8bf t0 = *(const v8bf*)(qrow + loA);
    v8bf t1 = *(const v8bf*)(qrow + loA + 8);
    v8bf t2 = *(const v8bf*)(qrow + 32 + loA);
    v8bf t3 = *(const v8bf*)(qrow + 32 + loA + 8);
#pragma unroll
    for (int i = 0; i < 8; ++i) {
      qa0[i] = t0[i]; qa0[8 + i] = t1[i];
      qa1[i] = t2[i]; qa1[8 + i] = t3[i];
    }
  }

  v8f o0 = {}, o1 = {}, o2 = {}, o3 = {};
  float rm[8], rl[8];
#pragma unroll
  for (int r = 0; r < 8; ++r) { rm[r] = -1e30f; rl[r] = 0.f; }

  const bf16* kbase = Kc + (size_t)b * SEQ * D_MODEL + h * HEAD_DIM;
  const bf16* vbase = Vt + (size_t)(b * N_HEADS + h) * HEAD_DIM * SEQ;

  const int nkb = (qt * 16 + 15) / 32 + 1;
  for (int kb = 0; kb < nkb; ++kb) {
    // ---- scores: S = Q (16x64) @ K_block^T (64x32), two 16-col halves
    const bf16* k0row = kbase + (size_t)(kb * 32 + row) * D_MODEL;
    const bf16* k1row = kbase + (size_t)(kb * 32 + 16 + row) * D_MODEL;
    v16bf kf00 = *(const v16bf*)(k0row + loB);
    v16bf kf01 = *(const v16bf*)(k0row + 32 + loB);
    v16bf kf10 = *(const v16bf*)(k1row + loB);
    v16bf kf11 = *(const v16bf*)(k1row + 32 + loB);

    v8f s0 = {}, s1 = {};
    s0 = __builtin_amdgcn_wmma_f32_16x16x32_bf16(false, qa0, false, kf00,
                                                 (short)0, s0, false, false);
    s0 = __builtin_amdgcn_wmma_f32_16x16x32_bf16(false, qa1, false, kf01,
                                                 (short)0, s0, false, false);
    s1 = __builtin_amdgcn_wmma_f32_16x16x32_bf16(false, qa0, false, kf10,
                                                 (short)0, s1, false, false);
    s1 = __builtin_amdgcn_wmma_f32_16x16x32_bf16(false, qa1, false, kf11,
                                                 (short)0, s1, false, false);

    // ---- scale + causal mask + online softmax
    const int n0 = kb * 32 + row;
    const int n1 = n0 + 16;
    float p0[8], p1[8];
#pragma unroll
    for (int r = 0; r < 8; ++r) {
      const int mg = qt * 16 + r + hi8;
      float x0 = (n0 <= mg) ? s0[r] * 0.125f : -1e30f;
      float x1 = (n1 <= mg) ? s1[r] * 0.125f : -1e30f;
      p0[r] = x0; p1[r] = x1;
      float mx = fmaxf(x0, x1);
#pragma unroll
      for (int m = 1; m < 16; m <<= 1)
        mx = fmaxf(mx, __shfl_xor(mx, m, 32));
      float nm = fmaxf(rm[r], mx);
      float scl = __expf(rm[r] - nm);
      rm[r] = nm;
      p0[r] = __expf(p0[r] - nm);
      p1[r] = __expf(p1[r] - nm);
      float ps = p0[r] + p1[r];
#pragma unroll
      for (int m = 1; m < 16; m <<= 1) ps += __shfl_xor(ps, m, 32);
      rl[r] = rl[r] * scl + ps;
      o0[r] *= scl; o1[r] *= scl; o2[r] *= scl; o3[r] *= scl;
    }

    // ---- P: C-layout -> A-layout via per-wave LDS bounce (bf16)
#pragma unroll
    for (int r = 0; r < 8; ++r) {
      pbuf[(r + hi8) * 32 + row]      = (bf16)p0[r];
      pbuf[(r + hi8) * 32 + row + 16] = (bf16)p1[r];
    }
    __syncthreads();  // single-wave block: orders DS store -> DS load
    v16bf pf;
    {
      v8bf t0 = *(const v8bf*)(pbuf + row * 32 + loA);
      v8bf t1 = *(const v8bf*)(pbuf + row * 32 + loA + 8);
#pragma unroll
      for (int i = 0; i < 8; ++i) { pf[i] = t0[i]; pf[8 + i] = t1[i]; }
    }
    __syncthreads();

    // ---- O += P (16x32) @ V_block (32x64); V pre-transposed: rows are d
    v16bf vf0 = *(const v16bf*)(vbase + (size_t)(row)      * SEQ + kb * 32 + loB);
    v16bf vf1 = *(const v16bf*)(vbase + (size_t)(16 + row) * SEQ + kb * 32 + loB);
    v16bf vf2 = *(const v16bf*)(vbase + (size_t)(32 + row) * SEQ + kb * 32 + loB);
    v16bf vf3 = *(const v16bf*)(vbase + (size_t)(48 + row) * SEQ + kb * 32 + loB);
    o0 = __builtin_amdgcn_wmma_f32_16x16x32_bf16(false, pf, false, vf0,
                                                 (short)0, o0, false, false);
    o1 = __builtin_amdgcn_wmma_f32_16x16x32_bf16(false, pf, false, vf1,
                                                 (short)0, o1, false, false);
    o2 = __builtin_amdgcn_wmma_f32_16x16x32_bf16(false, pf, false, vf2,
                                                 (short)0, o2, false, false);
    o3 = __builtin_amdgcn_wmma_f32_16x16x32_bf16(false, pf, false, vf3,
                                                 (short)0, o3, false, false);
  }

  // ---- normalize and emit context [b, s, h*64 + d] in bf16
#pragma unroll
  for (int r = 0; r < 8; ++r) {
    float inv = 1.0f / rl[r];
    size_t base =
        (size_t)(b * SEQ + qt * 16 + r + hi8) * D_MODEL + h * HEAD_DIM;
    Ctx[base + row]      = (bf16)(o0[r] * inv);
    Ctx[base + 16 + row] = (bf16)(o1[r] * inv);
    Ctx[base + 32 + row] = (bf16)(o2[r] * inv);
    Ctx[base + 48 + row] = (bf16)(o3[r] * inv);
  }
}

// ---------------------------------------------------------------- launch
extern "C" void kernel_launch(void* const* d_in, const int* in_sizes, int n_in,
                              void* d_out, int out_size, void* d_ws,
                              size_t ws_size, hipStream_t stream) {
  (void)in_sizes; (void)n_in; (void)out_size; (void)ws_size;
  const float* x  = (const float*)d_in[0];
  // d_in[1] = causal mask (int32) — causality applied analytically, unused.
  const float* Wq = (const float*)d_in[2];
  const float* Wk = (const float*)d_in[3];
  const float* Wv = (const float*)d_in[4];
  const float* Wo = (const float*)d_in[5];

  char* ws = (char*)d_ws;
  bf16* x16  = (bf16*)(ws + ((size_t)0  << 20));
  bf16* wq16 = (bf16*)(ws + ((size_t)8  << 20));
  bf16* wk16 = (bf16*)(ws + ((size_t)10 << 20));
  bf16* wv16 = (bf16*)(ws + ((size_t)12 << 20));
  bf16* wo16 = (bf16*)(ws + ((size_t)14 << 20));
  bf16* q16  = (bf16*)(ws + ((size_t)16 << 20));
  bf16* k16  = (bf16*)(ws + ((size_t)24 << 20));
  bf16* vT16 = (bf16*)(ws + ((size_t)32 << 20));
  bf16* c16  = (bf16*)(ws + ((size_t)40 << 20));

  const int NX = BATCH * SEQ * D_MODEL;   // 4 Mi elems
  const int NW = D_MODEL * D_MODEL;       // 1 Mi elems
  cvt_f32_bf16<<<1024, 256, 0, stream>>>(x,  x16,  NX);
  cvt_f32_bf16<<<512,  256, 0, stream>>>(Wq, wq16, NW);
  cvt_f32_bf16<<<512,  256, 0, stream>>>(Wk, wk16, NW);
  cvt_f32_bf16<<<512,  256, 0, stream>>>(Wv, wv16, NW);
  cvt_f32_bf16<<<512,  256, 0, stream>>>(Wo, wo16, NW);

  const int M = BATCH * SEQ;                        // 4096
  const int ntiles = (M / 64) * (D_MODEL / 64);     // 1024 waves (64x64 tiles)
  const int gblocks = ntiles / 4;                   // 128-thread blocks
  gemm_bf16_wmma<<<gblocks, 128, 0, stream>>>(x16, wq16, q16, M, D_MODEL, D_MODEL, 0);
  gemm_bf16_wmma<<<gblocks, 128, 0, stream>>>(x16, wk16, k16, M, D_MODEL, D_MODEL, 0);
  gemm_bf16_wmma<<<gblocks, 128, 0, stream>>>(x16, wv16, vT16, M, D_MODEL, D_MODEL, 1);

  const int awaves = BATCH * N_HEADS * (SEQ / 16);  // 4096 waves, 1/block
  attn_wmma<<<awaves, 32, 0, stream>>>(q16, k16, vT16, c16);

  gemm_bf16_wmma<<<gblocks, 128, 0, stream>>>(c16, wo16, d_out, M, D_MODEL, D_MODEL, 2);
}
